// Transformer_1082331758963
// MI455X (gfx1250) — matrix operand order repaired and checked by reading, not compiled, not run on hardware
//
#include <hip/hip_runtime.h>
#include <hip/hip_bf16.h>

// ---------------------------------------------------------------------------
// Model constants (match reference)
// ---------------------------------------------------------------------------
#define BSZ   2
#define CSRC  1024
#define CTGT  1024
#define EMB   512
#define DQK   64
#define DFFN  2048
#define NHEAD 8
#define NLAY  6
#define VOCAB 32000
#define BT    (BSZ * CSRC)          // 2048 rows (same for src & tgt)

typedef long long ll;
typedef __attribute__((ext_vector_type(16))) __bf16 bf16x16;
typedef __attribute__((ext_vector_type(8)))  __bf16 bf16x8;
typedef __attribute__((ext_vector_type(8)))  float  f32x8;

// ---------------------------------------------------------------------------
// bf16 helpers
// ---------------------------------------------------------------------------
__device__ __forceinline__ unsigned short f2bf(float f) {
  unsigned int u = __float_as_uint(f);
  unsigned int r = u + 0x7FFFu + ((u >> 16) & 1u);   // round-to-nearest-even
  return (unsigned short)(r >> 16);
}

__global__ void f32_to_bf16_kernel(const float* __restrict__ s,
                                   unsigned short* __restrict__ d, ll n) {
  ll i = (ll)blockIdx.x * 256 + threadIdx.x;
  ll stride = (ll)gridDim.x * 256;
  for (; i < n; i += stride) d[i] = f2bf(s[i]);
}

// ---------------------------------------------------------------------------
// Batched bf16 WMMA GEMM, double-buffered LDS staging with CDNA5 async
// global->LDS copies on the straight-copy orientations.
//   C[M,N] = alpha * (A op* B) + bias, optional ReLU, optional bf16 mirror.
//   TA: A given as [K,M] row-major;  TB: B given as [N,K] row-major.
// Block = 256 threads (8 wave32), block tile 128(M) x 64(N), K step 32.
// LDS: A tile 128x32 (K-contiguous rows), B tile stored transposed [64n][32k]
// so every fragment is a K-contiguous run:
//   A frag = two 16B ds loads (K koff..+7, koff+8..+15), contiguous K layout
//   B frag = one 32B ds load  (K kbase..+15),            contiguous K layout
// While wave computes on buffer b, buffer b^1 is filled with
// global_load_async_to_lds_b128 (ASYNCcnt) / manual staged stores.
// All shapes used divide the tile exactly (M%128==N%64==K%32==0).
// ---------------------------------------------------------------------------
template <int TA, int TB, int RELU>
__global__ __launch_bounds__(256)
void wmma_gemm_bf16(const unsigned short* __restrict__ A,
                    const unsigned short* __restrict__ Bm,
                    float* __restrict__ C,
                    unsigned short* __restrict__ Cb,
                    const float* __restrict__ bias,
                    int M, int N, int K,
                    int lda, int ldb, int ldc,
                    ll sA, ll sB, ll sC, float alpha) {
  __shared__ __attribute__((aligned(32))) unsigned short Ads[2][128 * 32];
  __shared__ __attribute__((aligned(32))) unsigned short Bds[2][64 * 32];
  (void)M; (void)N;

  A  += (ll)blockIdx.z * sA;
  Bm += (ll)blockIdx.z * sB;
  C  += (ll)blockIdx.z * sC;
  if (Cb) Cb += (ll)blockIdx.z * sC;

  const int tid  = threadIdx.x;
  const int wave = tid >> 5;
  const int lane = tid & 31;
  const int gRow = blockIdx.y * 128;
  const int gCol = blockIdx.x * 64;

  f32x8 acc0 = {}, acc1 = {}, acc2 = {}, acc3 = {};

  // per-lane fragment coordinates (constant over K loop)
  const int arowL = wave * 16 + (lane & 15);   // local M row for this lane
  const int akoff = (lane >> 4) * 16;          // A: hi half-wave -> K+16 block
  const int bkoff = (lane >> 4) * 16;          // B: hi half-wave -> K+16 block
  const int bcol  = lane & 15;

  // stage one 128x32 A tile + 32x64 B tile (as [n][k]) into buffer `buf`
  auto stage = [&](int k0, int buf) {
    unsigned short* Abuf = &Ads[buf][0];
    unsigned short* Bbuf = &Bds[buf][0];
    // ---------------- A tile ----------------
    if (!TA) {
      const int r = tid >> 1, seg = tid & 1;
      const unsigned short* gp = A + (ll)(gRow + r) * lda + k0 + seg * 16;
      const unsigned lo = (unsigned)(size_t)&Abuf[r * 32 + seg * 16];
      // 32 bytes/thread; offset: applies to both global and LDS addresses
      asm volatile("global_load_async_to_lds_b128 %0, %1, off\n\t"
                   "global_load_async_to_lds_b128 %0, %1, off offset:16"
                   :: "v"(lo), "v"(gp) : "memory");
    } else {
      const int kk = tid >> 3, mseg = tid & 7;
      const unsigned short* gp = A + (ll)(k0 + kk) * lda + gRow + mseg * 16;
      union { uint4 u; unsigned short s[8]; } v0, v1;
      v0.u = *(const uint4*)gp;
      v1.u = *(const uint4*)(gp + 8);
#pragma unroll
      for (int i = 0; i < 8; ++i) {
        Abuf[(mseg * 16 + i) * 32 + kk]     = v0.s[i];
        Abuf[(mseg * 16 + 8 + i) * 32 + kk] = v1.s[i];
      }
    }
    // ---------------- B tile -> Bds[n][k] ----------------
    if (TB) {
      const int n = tid >> 2, kseg = tid & 3;
      const unsigned short* gp = Bm + (ll)(gCol + n) * ldb + k0 + kseg * 8;
      const unsigned lo = (unsigned)(size_t)&Bbuf[n * 32 + kseg * 8];
      asm volatile("global_load_async_to_lds_b128 %0, %1, off"
                   :: "v"(lo), "v"(gp) : "memory");
    } else {
      const int kk = tid >> 3, nseg = tid & 7;
      const unsigned short* gp = Bm + (ll)(k0 + kk) * ldb + gCol + nseg * 8;
      union { uint4 u; unsigned short s[8]; } v;
      v.u = *(const uint4*)gp;
#pragma unroll
      for (int i = 0; i < 8; ++i) Bbuf[(nseg * 8 + i) * 32 + kk] = v.s[i];
    }
  };

  stage(0, 0);
  if (!TA || TB) asm volatile("s_wait_asynccnt 0x0" ::: "memory");
  __syncthreads();

  int buf = 0;
  for (int k0 = 0; k0 < K; k0 += 32) {
    // prefetch next tile into the other buffer while we compute
    if (k0 + 32 < K) stage(k0 + 32, buf ^ 1);

    // ---------------- fragments + WMMA on current buffer ----------------
    const bf16x8 alo = *(const bf16x8*)&Ads[buf][arowL * 32 + akoff];
    const bf16x8 ahi = *(const bf16x8*)&Ads[buf][arowL * 32 + akoff + 8];
    const bf16x16 af = __builtin_shufflevector(
        alo, ahi, 0, 1, 2, 3, 4, 5, 6, 7, 8, 9, 10, 11, 12, 13, 14, 15);

    {
      const bf16x16 bf = *(const bf16x16*)&Bds[buf][(0 * 16 + bcol) * 32 + bkoff];
      acc0 = __builtin_amdgcn_wmma_f32_16x16x32_bf16(false, af, false, bf,
                                                     (short)0, acc0, false, false);
    }
    {
      const bf16x16 bf = *(const bf16x16*)&Bds[buf][(1 * 16 + bcol) * 32 + bkoff];
      acc1 = __builtin_amdgcn_wmma_f32_16x16x32_bf16(false, af, false, bf,
                                                     (short)0, acc1, false, false);
    }
    {
      const bf16x16 bf = *(const bf16x16*)&Bds[buf][(2 * 16 + bcol) * 32 + bkoff];
      acc2 = __builtin_amdgcn_wmma_f32_16x16x32_bf16(false, af, false, bf,
                                                     (short)0, acc2, false, false);
    }
    {
      const bf16x16 bf = *(const bf16x16*)&Bds[buf][(3 * 16 + bcol) * 32 + bkoff];
      acc3 = __builtin_amdgcn_wmma_f32_16x16x32_bf16(false, af, false, bf,
                                                     (short)0, acc3, false, false);
    }

    // async writes into buf^1 must land before anyone reads them (and before
    // anyone overwrites buf next iteration)
    if (!TA || TB) asm volatile("s_wait_asynccnt 0x0" ::: "memory");
    __syncthreads();
    buf ^= 1;
  }

  // ---- epilogue: alpha, bias, relu, f32 store (+ optional bf16 mirror) ----
  const int mrow0 = gRow + wave * 16 + ((lane >> 4) * 8);
#pragma unroll
  for (int nt = 0; nt < 4; ++nt) {
    const int col = gCol + nt * 16 + bcol;
    const float bv = bias ? bias[col] : 0.0f;
    const f32x8 acc = (nt == 0) ? acc0 : (nt == 1) ? acc1 : (nt == 2) ? acc2 : acc3;
#pragma unroll
    for (int r = 0; r < 8; ++r) {
      const int m = mrow0 + r;
      float v = acc[r] * alpha + bv;
      if (RELU) v = fmaxf(v, 0.0f);
      C[(ll)m * ldc + col] = v;
      if (Cb) Cb[(ll)m * ldc + col] = f2bf(v);
    }
  }
}

// ---------------------------------------------------------------------------
// Softmax over the KEY axis (axis stride = T in the [S,T] score matrix),
// analytic causal mask (key s > query t masked). Writes bf16 probabilities.
// Consecutive threads walk consecutive t -> coalesced at every key step.
// ---------------------------------------------------------------------------
__global__ void softmax_key_kernel(const float* __restrict__ s,
                                   unsigned short* __restrict__ p,
                                   int S, int T, int causal) {
  const ll base = (ll)blockIdx.y * (ll)S * T;
  const int t = blockIdx.x * blockDim.x + threadIdx.x;
  const int limit = causal ? (t + 1) : S;
  const float* sc = s + base + t;
  unsigned short* pc = p + base + t;

  float m = -3.4e38f;
  for (int i = 0; i < limit; ++i) m = fmaxf(m, sc[(ll)i * T]);
  float sum = 0.0f;
  for (int i = 0; i < limit; ++i) sum += __expf(sc[(ll)i * T] - m);
  const float inv = 1.0f / sum;
  for (int i = 0; i < limit; ++i) pc[(ll)i * T] = f2bf(__expf(sc[(ll)i * T] - m) * inv);
  for (int i = limit; i < S; ++i) pc[(ll)i * T] = 0;
}

// ---------------------------------------------------------------------------
// Fused residual (+ optional 8-head sum) + LayerNorm. One block per row
// (E = 512, 256 threads -> 2 elems/thread). Emits f32 + bf16.
// ---------------------------------------------------------------------------
__global__ __launch_bounds__(256)
void ln_fused_kernel(const float* __restrict__ x,
                     const float* __restrict__ add, int nheads, ll headStride,
                     const float* __restrict__ g, const float* __restrict__ b,
                     float* __restrict__ xo, unsigned short* __restrict__ xob) {
  __shared__ float red[256];
  const int r = blockIdx.x, tid = threadIdx.x;
  const ll ro = (ll)r * EMB;
  float v0 = x[ro + tid], v1 = x[ro + tid + 256];
  if (nheads > 0) {
    for (int h = 0; h < nheads; ++h) {
      const float* a = add + (ll)h * headStride + ro;
      v0 += a[tid];
      v1 += a[tid + 256];
    }
  } else {
    v0 += add[ro + tid];
    v1 += add[ro + tid + 256];
  }
  red[tid] = v0 + v1;
  __syncthreads();
  for (int st = 128; st > 0; st >>= 1) {
    if (tid < st) red[tid] += red[tid + st];
    __syncthreads();
  }
  const float mu = red[0] * (1.0f / EMB);
  __syncthreads();
  red[tid] = (v0 - mu) * (v0 - mu) + (v1 - mu) * (v1 - mu);
  __syncthreads();
  for (int st = 128; st > 0; st >>= 1) {
    if (tid < st) red[tid] += red[tid + st];
    __syncthreads();
  }
  const float rstd = rsqrtf(red[0] * (1.0f / EMB) + 1e-5f);
  const float o0 = (v0 - mu) * rstd * g[tid] + b[tid];
  const float o1 = (v1 - mu) * rstd * g[tid + 256] + b[tid + 256];
  xo[ro + tid] = o0;
  xo[ro + tid + 256] = o1;
  xob[ro + tid] = f2bf(o0);
  xob[ro + tid + 256] = f2bf(o1);
}

// ---------------------------------------------------------------------------
// Embedding * sqrt(E) + sinusoidal positional encoding. One block per row.
// ---------------------------------------------------------------------------
__global__ __launch_bounds__(256)
void embed_kernel(const int* __restrict__ tok, const float* __restrict__ emb,
                  float* __restrict__ x, unsigned short* __restrict__ xb,
                  int seqlen, float sq) {
  const int r = blockIdx.x;
  const int pos = r % seqlen;
  const int tokid = tok[r];
  const ll ro = (ll)r * EMB;
  const ll to = (ll)tokid * EMB;
#pragma unroll
  for (int j = 0; j < 2; ++j) {
    const int e = threadIdx.x + j * 256;
    const int i = e >> 1;
    const float div = __expf(-(float)(2 * i) * (logf(10000.0f) / (float)EMB));
    const float ang = (float)pos * div;
    const float pe = (e & 1) ? __cosf(ang) : __sinf(ang);
    const float v = emb[to + e] * sq + pe;
    x[ro + e] = v;
    xb[ro + e] = f2bf(v);
  }
}

// ---------------------------------------------------------------------------
// In-place row log-softmax over V = 32000 (contiguous last axis).
// ---------------------------------------------------------------------------
__global__ __launch_bounds__(256)
void log_softmax_kernel(float* __restrict__ x, int V) {
  __shared__ float red[256];
  float* row = x + (ll)blockIdx.x * V;
  const int tid = threadIdx.x;
  float m = -3.4e38f;
  for (int i = tid; i < V; i += 256) m = fmaxf(m, row[i]);
  red[tid] = m;
  __syncthreads();
  for (int st = 128; st > 0; st >>= 1) {
    if (tid < st) red[tid] = fmaxf(red[tid], red[tid + st]);
    __syncthreads();
  }
  m = red[0];
  __syncthreads();
  float sum = 0.0f;
  for (int i = tid; i < V; i += 256) sum += __expf(row[i] - m);
  red[tid] = sum;
  __syncthreads();
  for (int st = 128; st > 0; st >>= 1) {
    if (tid < st) red[tid] += red[tid + st];
    __syncthreads();
  }
  const float lz = m + logf(red[0]);
  for (int i = tid; i < V; i += 256) row[i] -= lz;
}

// ---------------------------------------------------------------------------
// Host-side orchestration
// ---------------------------------------------------------------------------
static void gemm_disp(hipStream_t stream,
                      const unsigned short* A, const unsigned short* Bm,
                      float* C, unsigned short* Cb, const float* bias,
                      int M, int N, int K, int lda, int ldb, int ldc,
                      ll sA, ll sB, ll sC, int batch, float alpha,
                      int ta, int tb, int relu) {
  dim3 g((unsigned)(N / 64), (unsigned)(M / 128), (unsigned)batch), blk(256);
  if (!ta && !tb && !relu)
    wmma_gemm_bf16<0, 0, 0><<<g, blk, 0, stream>>>(A, Bm, C, Cb, bias, M, N, K, lda, ldb, ldc, sA, sB, sC, alpha);
  else if (!ta && !tb && relu)
    wmma_gemm_bf16<0, 0, 1><<<g, blk, 0, stream>>>(A, Bm, C, Cb, bias, M, N, K, lda, ldb, ldc, sA, sB, sC, alpha);
  else if (!ta && tb)
    wmma_gemm_bf16<0, 1, 0><<<g, blk, 0, stream>>>(A, Bm, C, Cb, bias, M, N, K, lda, ldb, ldc, sA, sB, sC, alpha);
  else
    wmma_gemm_bf16<1, 0, 0><<<g, blk, 0, stream>>>(A, Bm, C, Cb, bias, M, N, K, lda, ldb, ldc, sA, sB, sC, alpha);
}

extern "C" void kernel_launch(void* const* d_in, const int* in_sizes, int n_in,
                              void* d_out, int out_size, void* d_ws, size_t ws_size,
                              hipStream_t stream) {
  (void)in_sizes; (void)n_in; (void)out_size; (void)ws_size;
  // Input order (flattened setup_inputs dict):
  const float* emb_src = (const float*)d_in[0];
  const float* emb_tgt = (const float*)d_in[1];
  const float* e_wq = (const float*)d_in[2];   // [L,H,E,DQ]
  const float* e_wk = (const float*)d_in[3];
  const float* e_wv = (const float*)d_in[4];   // [L,H,E,E]
  const float* e_w1 = (const float*)d_in[5];   // [L,E,DFF]
  const float* e_b1 = (const float*)d_in[6];
  const float* e_w2 = (const float*)d_in[7];   // [L,DFF,E]
  const float* e_b2 = (const float*)d_in[8];
  const float* e_lng = (const float*)d_in[9];  // [L,2,E]
  const float* e_lnb = (const float*)d_in[10];
  const float* d_wq_sa = (const float*)d_in[11];
  const float* d_wk_sa = (const float*)d_in[12];
  const float* d_wv_sa = (const float*)d_in[13];
  const float* d_wq_ca = (const float*)d_in[14];
  const float* d_wk_ca = (const float*)d_in[15];
  const float* d_wv_ca = (const float*)d_in[16];
  const float* d_w1 = (const float*)d_in[17];
  const float* d_b1 = (const float*)d_in[18];
  const float* d_w2 = (const float*)d_in[19];
  const float* d_b2 = (const float*)d_in[20];
  const float* d_lng = (const float*)d_in[21]; // [L,3,E]
  const float* d_lnb = (const float*)d_in[22];
  const float* gen_w = (const float*)d_in[23]; // [E,V]
  const float* gen_b = (const float*)d_in[24];
  const int* src = (const int*)d_in[25];
  const int* tgt = (const int*)d_in[26];
  // d_in[27..29] = masks (all-false / analytic causal -> realized in-kernel)

  // ---- bump allocator over d_ws ----
  size_t off = 0;
  auto alloc = [&](ll bytes) -> char* {
    size_t a = (off + 255) & ~(size_t)255;
    off = a + (size_t)bytes;
    return (char*)d_ws + a;
  };
  auto toBf = [&](const float* s, ll n) -> unsigned short* {
    unsigned short* d = (unsigned short*)alloc(n * 2);
    ll blocks = (n + 255) / 256;
    if (blocks > 8192) blocks = 8192;
    f32_to_bf16_kernel<<<dim3((unsigned)blocks), dim3(256), 0, stream>>>(s, d, n);
    return d;
  };

  // ---- weight conversion to bf16 (per call; deterministic) ----
  const ll nQK = (ll)NLAY * NHEAD * EMB * DQK;
  const ll nV  = (ll)NLAY * NHEAD * EMB * EMB;
  const ll nW1 = (ll)NLAY * EMB * DFFN;
  unsigned short* eWq = toBf(e_wq, nQK);
  unsigned short* eWk = toBf(e_wk, nQK);
  unsigned short* eWv = toBf(e_wv, nV);
  unsigned short* eW1 = toBf(e_w1, nW1);
  unsigned short* eW2 = toBf(e_w2, nW1);
  unsigned short* dWqS = toBf(d_wq_sa, nQK);
  unsigned short* dWkS = toBf(d_wk_sa, nQK);
  unsigned short* dWvS = toBf(d_wv_sa, nV);
  unsigned short* dWqC = toBf(d_wq_ca, nQK);
  unsigned short* dWkC = toBf(d_wk_ca, nQK);
  unsigned short* dWvC = toBf(d_wv_ca, nV);
  unsigned short* dW1 = toBf(d_w1, nW1);
  unsigned short* dW2 = toBf(d_w2, nW1);
  unsigned short* genWb = toBf(gen_w, (ll)EMB * VOCAB);

  // ---- activation buffers ----
  const int T = CSRC;                  // == CTGT
  float* xe_f = (float*)alloc((ll)BT * EMB * 4);
  unsigned short* xe_b = (unsigned short*)alloc((ll)BT * EMB * 2);
  float* xd_f = (float*)alloc((ll)BT * EMB * 4);
  unsigned short* xd_b = (unsigned short*)alloc((ll)BT * EMB * 2);
  float* Qf = (float*)alloc((ll)NHEAD * BT * DQK * 4);
  unsigned short* Qb = (unsigned short*)alloc((ll)NHEAD * BT * DQK * 2);
  float* Kf = (float*)alloc((ll)NHEAD * BT * DQK * 4);
  unsigned short* Kb = (unsigned short*)alloc((ll)NHEAD * BT * DQK * 2);
  float* Vf = (float*)alloc((ll)NHEAD * BT * EMB * 4);
  unsigned short* Vb = (unsigned short*)alloc((ll)NHEAD * BT * EMB * 2);
  float* scores = (float*)alloc((ll)NHEAD * BSZ * T * T * 4);
  unsigned short* p_b = (unsigned short*)alloc((ll)NHEAD * BSZ * T * T * 2);
  float* headout = (float*)alloc((ll)NHEAD * BT * EMB * 4);
  float* hid = (float*)alloc((ll)BT * DFFN * 4);
  unsigned short* hid_b = (unsigned short*)alloc((ll)BT * DFFN * 2);
  float* ffnout = (float*)alloc((ll)BT * EMB * 4);

  const float invSqDq = 0.125f;  // 1/sqrt(64)

  // MHA: x_f/x_b = LN(x + sum_h head_h(attn(xq, xkv)))
  auto attention = [&](const unsigned short* xq_b, const unsigned short* xkv_b,
                       const unsigned short* wq, const unsigned short* wk,
                       const unsigned short* wv, int causal,
                       float* x_f, unsigned short* x_b,
                       const float* lng, const float* lnb) {
    // Q/K/V projections, batched over heads
    gemm_disp(stream, xq_b, wq, Qf, Qb, nullptr, BT, DQK, EMB, EMB, DQK, DQK,
              0, (ll)EMB * DQK, (ll)BT * DQK, NHEAD, 1.0f, 0, 0, 0);
    gemm_disp(stream, xkv_b, wk, Kf, Kb, nullptr, BT, DQK, EMB, EMB, DQK, DQK,
              0, (ll)EMB * DQK, (ll)BT * DQK, NHEAD, 1.0f, 0, 0, 0);
    gemm_disp(stream, xkv_b, wv, Vf, Vb, nullptr, BT, EMB, EMB, EMB, EMB, EMB,
              0, (ll)EMB * EMB, (ll)BT * EMB, NHEAD, 1.0f, 0, 0, 0);
    // scores[s,t] = K[s,:] . Q[t,:] / sqrt(dq)  (B = Q, transposed)
    gemm_disp(stream, Kb, Qb, scores, nullptr, nullptr, T, T, DQK, DQK, DQK, T,
              (ll)T * DQK, (ll)T * DQK, (ll)T * T, NHEAD * BSZ, invSqDq, 0, 1, 0);
    softmax_key_kernel<<<dim3(T / 256, NHEAD * BSZ), dim3(256), 0, stream>>>(
        scores, p_b, T, T, causal);
    // head out[t,f] = sum_s p[s,t] * V[s,f]  (A = p, transposed)
    gemm_disp(stream, p_b, Vb, headout, nullptr, nullptr, T, EMB, T, T, EMB, EMB,
              (ll)T * T, (ll)T * EMB, (ll)T * EMB, NHEAD * BSZ, 1.0f, 1, 0, 0);
    ln_fused_kernel<<<dim3(BT), dim3(256), 0, stream>>>(
        x_f, headout, NHEAD, (ll)BT * EMB, lng, lnb, x_f, x_b);
  };

  auto ffn = [&](float* x_f, unsigned short* x_b,
                 const unsigned short* w1, const float* b1,
                 const unsigned short* w2, const float* b2,
                 const float* lng, const float* lnb) {
    gemm_disp(stream, x_b, w1, hid, hid_b, b1, BT, DFFN, EMB, EMB, DFFN, DFFN,
              0, 0, 0, 1, 1.0f, 0, 0, 1);   // + bias, ReLU, bf16 mirror
    gemm_disp(stream, hid_b, w2, ffnout, nullptr, b2, BT, EMB, DFFN, DFFN, EMB, EMB,
              0, 0, 0, 1, 1.0f, 0, 0, 0);
    ln_fused_kernel<<<dim3(BT), dim3(256), 0, stream>>>(
        x_f, ffnout, 0, 0, lng, lnb, x_f, x_b);
  };

  const float sq = sqrtf((float)EMB);

  // ------------------- Encoder -------------------
  embed_kernel<<<dim3(BT), dim3(256), 0, stream>>>(src, emb_src, xe_f, xe_b, CSRC, sq);
  for (int l = 0; l < NLAY; ++l) {
    attention(xe_b, xe_b,
              eWq + (ll)l * NHEAD * EMB * DQK, eWk + (ll)l * NHEAD * EMB * DQK,
              eWv + (ll)l * NHEAD * EMB * EMB, /*causal=*/0,
              xe_f, xe_b, e_lng + (ll)(l * 2 + 0) * EMB, e_lnb + (ll)(l * 2 + 0) * EMB);
    ffn(xe_f, xe_b, eW1 + (ll)l * EMB * DFFN, e_b1 + (ll)l * DFFN,
        eW2 + (ll)l * DFFN * EMB, e_b2 + (ll)l * EMB,
        e_lng + (ll)(l * 2 + 1) * EMB, e_lnb + (ll)(l * 2 + 1) * EMB);
  }
  // xe_b now holds bf16 encoder memory

  // ------------------- Decoder -------------------
  embed_kernel<<<dim3(BT), dim3(256), 0, stream>>>(tgt, emb_tgt, xd_f, xd_b, CTGT, sq);
  for (int l = 0; l < NLAY; ++l) {
    attention(xd_b, xd_b,
              dWqS + (ll)l * NHEAD * EMB * DQK, dWkS + (ll)l * NHEAD * EMB * DQK,
              dWvS + (ll)l * NHEAD * EMB * EMB, /*causal=*/1,
              xd_f, xd_b, d_lng + (ll)(l * 3 + 0) * EMB, d_lnb + (ll)(l * 3 + 0) * EMB);
    attention(xd_b, xe_b,
              dWqC + (ll)l * NHEAD * EMB * DQK, dWkC + (ll)l * NHEAD * EMB * DQK,
              dWvC + (ll)l * NHEAD * EMB * EMB, /*causal=*/0,
              xd_f, xd_b, d_lng + (ll)(l * 3 + 1) * EMB, d_lnb + (ll)(l * 3 + 1) * EMB);
    ffn(xd_f, xd_b, dW1 + (ll)l * EMB * DFFN, d_b1 + (ll)l * DFFN,
        dW2 + (ll)l * DFFN * EMB, d_b2 + (ll)l * EMB,
        d_lng + (ll)(l * 3 + 2) * EMB, d_lnb + (ll)(l * 3 + 2) * EMB);
  }

  // ------------------- Generator + log-softmax -------------------
  float* logits = (float*)d_out;   // [B*T, V]
  gemm_disp(stream, xd_b, genWb, logits, nullptr, gen_b, BT, VOCAB, EMB,
            EMB, VOCAB, VOCAB, 0, 0, 0, 1, 1.0f, 0, 0, 0);
  log_softmax_kernel<<<dim3(BT), dim3(256), 0, stream>>>(logits, VOCAB);
}
